// EnhancedAttention_81793357185231
// MI455X (gfx1250) — matrix operand order repaired and checked by reading, not compiled, not run on hardware
//
#include <hip/hip_runtime.h>
#include <hip/hip_bf16.h>

#define D_MODEL 1024
#define S_LEN   2048
#define BATCH   2
#define NH      16
#define HD      64

typedef __attribute__((ext_vector_type(16))) _Float16 v16h;
typedef __attribute__((ext_vector_type(8)))  _Float16 v8h;
typedef __attribute__((ext_vector_type(8)))  float    v8f;

// D = A(16x32 f16) * B(32x16 f16) + C(16x16 f32)
__device__ __forceinline__ v8f wmma16x16x32(v16h a, v16h b, v8f c) {
  return __builtin_amdgcn_wmma_f32_16x16x32_f16(
      /*neg_a=*/false, a, /*neg_b=*/false, b,
      /*c_mod=*/(short)0, c, /*reuse_a=*/false, /*reuse_b=*/false);
}

// A-fragment (16x32, MxK) from LDS tile stored row-major [M][K] (lda halfs).
// Contiguous k order (same as B): lane l holds row M=l&15; K = 16*(l>=16) + {0..15}.
__device__ __forceinline__ v16h load_frag_a(const _Float16* sm, int lda) {
  int lane = threadIdx.x & 31;
  int m  = lane & 15;
  int kb = (lane >> 4) << 4;
  const _Float16* p = sm + m * lda + kb;
  v8h lo = *(const v8h*)(p);
  v8h hi = *(const v8h*)(p + 8);
  v16h r;
#pragma unroll
  for (int i = 0; i < 8; ++i) { r[i] = lo[i]; r[i + 8] = hi[i]; }
  return r;
}

// B-fragment (32x16, KxN) from LDS stored as [N][K] (ldb halfs, K contiguous).
// Contiguous k order (same as A): lane l holds col N=l&15; K = 16*(l>=16) + {0..15} -> one v16h load.
__device__ __forceinline__ v16h load_frag_b(const _Float16* sm, int ldb) {
  int lane = threadIdx.x & 31;
  int n  = lane & 15;
  int kb = (lane >> 4) << 4;
  return *(const v16h*)(sm + n * ldb + kb);
}

// ---------------------------------------------------------------------------
// Kernel 1: Y = X * W^T + bias -> f16 [B,H,S,hd] (optionally scaled).
// Block tile 128(M) x 128(N), K-step 64, double-buffered LDS, 8 waves:
// wave = (mw in {0,32,64,96}: two 16-row subtiles) x (nw in {0,64}).
// Each wave: 8 accumulators, 16 WMMAs/stage; each B-fragment pair feeds 4
// WMMAs so LDS traffic is 1.5 b128 per WMMA.
// ---------------------------------------------------------------------------
__global__ __launch_bounds__(256) void qkv_gemm_kernel(
    const float* __restrict__ X, const float* __restrict__ W,
    const float* __restrict__ bias, _Float16* __restrict__ out,
    float out_scale) {
  __shared__ __align__(32) _Float16 As[2][128 * 64];
  __shared__ __align__(32) _Float16 Bs[2][128 * 64];

  const int m0 = blockIdx.x * 128;
  const int n0 = blockIdx.y * 128;
  const int t = threadIdx.x;
  const int wave = t >> 5, lane = t & 31;
  const int mw = (wave & 3) * 32;
  const int nw = (wave >> 2) * 64;

  const int sr = t >> 1, sc = (t & 1) * 32;  // staging: 128x64, 32 elems/thread

  _Float16 a16[32];
  _Float16 b16[32];

  auto load_regs = [&](int k0) {
    const float* ap = X + (size_t)(m0 + sr) * D_MODEL + k0 + sc;
    const float* bp = W + (size_t)(n0 + sr) * D_MODEL + k0 + sc;
#pragma unroll
    for (int i = 0; i < 32; ++i) a16[i] = (_Float16)ap[i];
#pragma unroll
    for (int i = 0; i < 32; ++i) b16[i] = (_Float16)bp[i];
  };
  auto store_stage = [&](int buf) {
    _Float16* ad = As[buf] + sr * 64 + sc;
    _Float16* bd = Bs[buf] + sr * 64 + sc;
#pragma unroll
    for (int i = 0; i < 32; ++i) ad[i] = a16[i];
#pragma unroll
    for (int i = 0; i < 32; ++i) bd[i] = b16[i];
  };

  v8f acc[2][4] = {{{}, {}, {}, {}}, {{}, {}, {}, {}}};

  const int NS = D_MODEL / 64;  // 16 stages
  load_regs(0);
  store_stage(0);
  for (int s = 0; s < NS; ++s) {
    __syncthreads();
    if (s + 1 < NS) load_regs((s + 1) * 64);  // overlap HBM/L2 with WMMA

    const _Float16* Ab = As[s & 1] + mw * 64;
    v16h fa[2][2];
#pragma unroll
    for (int m = 0; m < 2; ++m) {
      fa[m][0] = load_frag_a(Ab + (16 * m) * 64, 64);
      fa[m][1] = load_frag_a(Ab + (16 * m) * 64 + 32, 64);
    }
#pragma unroll
    for (int j = 0; j < 4; ++j) {
      const _Float16* Bb = Bs[s & 1] + (nw + 16 * j) * 64;
      v16h fb0 = load_frag_b(Bb, 64);
      v16h fb1 = load_frag_b(Bb + 32, 64);
#pragma unroll
      for (int m = 0; m < 2; ++m) {
        acc[m][j] = wmma16x16x32(fa[m][0], fb0, acc[m][j]);
        acc[m][j] = wmma16x16x32(fa[m][1], fb1, acc[m][j]);
      }
    }
    if (s + 1 < NS) store_stage((s + 1) & 1);
  }

  const int hl = lane >> 4;
  const int nn = lane & 15;
#pragma unroll
  for (int m = 0; m < 2; ++m) {
#pragma unroll
    for (int j = 0; j < 4; ++j) {
#pragma unroll
      for (int r = 0; r < 8; ++r) {
        int grow = m0 + mw + 16 * m + r + 8 * hl;  // b*S + s
        int gcol = n0 + nw + 16 * j + nn;          // h*hd + d
        float y = (acc[m][j][r] + bias[gcol]) * out_scale;
        int b = grow >> 11, s = grow & (S_LEN - 1);
        int h = gcol >> 6,  d = gcol & (HD - 1);
        out[(((size_t)b * NH + h) * S_LEN + s) * HD + d] = (_Float16)y;
      }
    }
  }
}

// ---------------------------------------------------------------------------
// Kernel 2: flash attention, 128-row q tile, 8 waves (16 q rows/wave).
// Q pre-scaled by 1/sqrt(hd). rel [H,S,S] / mask [B,S] streamed non-temporal.
// ---------------------------------------------------------------------------
__global__ __launch_bounds__(256) void attn_flash_kernel(
    const _Float16* __restrict__ Q, const _Float16* __restrict__ K,
    const _Float16* __restrict__ V, const float* __restrict__ rel,
    const float* __restrict__ mask, _Float16* __restrict__ ctx) {
  __shared__ __align__(32) _Float16 Qs[128 * 64];
  __shared__ __align__(32) _Float16 Ks[64 * 64];
  __shared__ __align__(32) _Float16 Vt[64 * 64];  // transposed: [d][kk]
  __shared__ __align__(32) _Float16 Ps[128 * 64];

  const int q0 = blockIdx.x * 128;
  const int h  = blockIdx.y;
  const int b  = blockIdx.z;
  const int t = threadIdx.x;
  const int wave = t >> 5, lane = t & 31;
  const int qw = wave * 16;

  const _Float16* Qb = Q + (((size_t)b * NH + h) * S_LEN + q0) * HD;
  const _Float16* Kb = K + (((size_t)b * NH + h) * S_LEN) * HD;
  const _Float16* Vb = V + (((size_t)b * NH + h) * S_LEN) * HD;

  {  // stage Q tile once (128x64)
    int r = t >> 1, c = (t & 1) * 32;
#pragma unroll
    for (int i = 0; i < 4; ++i)
      *(v8h*)(Qs + r * 64 + c + i * 8) =
          *(const v8h*)(Qb + (size_t)r * HD + c + i * 8);
  }
  __syncthreads();

  const v16h fq0 = load_frag_a(Qs + qw * 64, 64);
  const v16h fq1 = load_frag_a(Qs + qw * 64 + 32, 64);

  v8f O[4] = {{}, {}, {}, {}};
  float mrow[8], lrow[8];
#pragma unroll
  for (int r = 0; r < 8; ++r) { mrow[r] = -1e30f; lrow[r] = 0.f; }

  const int hl = lane >> 4;
  const int nn = lane & 15;
  const int kr = t >> 2, kc = (t & 3) * 16;  // K/V staging: 64x64, 16/thread

  for (int k0 = 0; k0 < S_LEN; k0 += 64) {
    // Issue the HBM-bound rel/mask stream first (non-temporal) so its latency
    // overlaps K/V staging and the QK^T WMMAs below.
    float rl[4][8], mk[4];
#pragma unroll
    for (int j = 0; j < 4; ++j) {
      int cg = k0 + 16 * j + nn;
      mk[j] = __builtin_nontemporal_load(mask + (size_t)b * S_LEN + cg);
#pragma unroll
      for (int r = 0; r < 8; ++r) {
        int rg = q0 + qw + r + 8 * hl;
        rl[j][r] = __builtin_nontemporal_load(
            rel + ((size_t)h * S_LEN + rg) * S_LEN + cg);
      }
    }

    __syncthreads();
    {  // stage K tile (row-major) and V tile transposed
      const _Float16* ks = Kb + (size_t)(k0 + kr) * HD + kc;
      const _Float16* vs = Vb + (size_t)(k0 + kr) * HD + kc;
#pragma unroll
      for (int i = 0; i < 2; ++i)
        *(v8h*)(Ks + kr * 64 + kc + i * 8) = *(const v8h*)(ks + i * 8);
#pragma unroll
      for (int i = 0; i < 2; ++i) {
        v8h vv = *(const v8h*)(vs + i * 8);
#pragma unroll
        for (int u = 0; u < 8; ++u) Vt[(kc + i * 8 + u) * 64 + kr] = vv[u];
      }
      if (k0 + 64 < S_LEN)  // global_prefetch_b8 of next K tile
        __builtin_prefetch((const char*)(Kb + (size_t)(k0 + 64 + kr) * HD + kc), 0, 0);
    }
    __syncthreads();

    // S = Q K^T (Q pre-scaled)
    v8f Sv[4];
#pragma unroll
    for (int j = 0; j < 4; ++j) {
      v8f z = {};
      v16h kb0 = load_frag_b(Ks + (16 * j) * 64, 64);
      v16h kb1 = load_frag_b(Ks + (16 * j) * 64 + 32, 64);
      z = wmma16x16x32(fq0, kb0, z);
      Sv[j] = wmma16x16x32(fq1, kb1, z);
    }

    // + relative_position + attention_mask
#pragma unroll
    for (int j = 0; j < 4; ++j)
#pragma unroll
      for (int r = 0; r < 8; ++r) Sv[j][r] += rl[j][r] + mk[j];

    // online softmax: row max / rescale / exp / row sum
    float mnew[8], alpha[8];
#pragma unroll
    for (int r = 0; r < 8; ++r) {
      float mx = fmaxf(fmaxf(Sv[0][r], Sv[1][r]), fmaxf(Sv[2][r], Sv[3][r]));
#pragma unroll
      for (int d = 1; d < 16; d <<= 1) mx = fmaxf(mx, __shfl_xor(mx, d, 32));
      mnew[r]  = fmaxf(mrow[r], mx);
      alpha[r] = __expf(mrow[r] - mnew[r]);
      mrow[r]  = mnew[r];
    }
#pragma unroll
    for (int r = 0; r < 8; ++r) {
      float sum = 0.f;
#pragma unroll
      for (int j = 0; j < 4; ++j) {
        float p = __expf(Sv[j][r] - mnew[r]);
        Sv[j][r] = p;
        sum += p;
      }
#pragma unroll
      for (int d = 1; d < 16; d <<= 1) sum += __shfl_xor(sum, d, 32);
      lrow[r] = lrow[r] * alpha[r] + sum;
    }

    // P (f16) -> LDS [q][kk]; same-wave DS ops are in-order, so reading our
    // own rows back as A-fragments needs no barrier.
#pragma unroll
    for (int j = 0; j < 4; ++j)
#pragma unroll
      for (int r = 0; r < 8; ++r) {
        Ps[(qw + r + 8 * hl) * 64 + 16 * j + nn] = (_Float16)Sv[j][r];
        O[j][r] *= alpha[r];
      }

    // O += P @ V
    v16h fp0 = load_frag_a(Ps + qw * 64, 64);
    v16h fp1 = load_frag_a(Ps + qw * 64 + 32, 64);
#pragma unroll
    for (int j = 0; j < 4; ++j) {
      v16h vb0 = load_frag_b(Vt + (16 * j) * 64, 64);
      v16h vb1 = load_frag_b(Vt + (16 * j) * 64 + 32, 64);
      O[j] = wmma16x16x32(fp0, vb0, O[j]);
      O[j] = wmma16x16x32(fp1, vb1, O[j]);
    }
  }

  // finalize: O / l -> ctx [B,S,D] f16 (head h occupies cols h*64..)
#pragma unroll
  for (int j = 0; j < 4; ++j) {
#pragma unroll
    for (int r = 0; r < 8; ++r) {
      float o = O[j][r] / lrow[r];
      int sg = q0 + qw + r + 8 * hl;
      int dd = h * HD + 16 * j + nn;
      __builtin_nontemporal_store(
          (_Float16)o, ctx + ((size_t)b * S_LEN + sg) * D_MODEL + dd);
    }
  }
}

// ---------------------------------------------------------------------------
// Kernel 3: out = ctx(f16) * Wo^T + bo (f32). Same tiling as kernel 1.
// ---------------------------------------------------------------------------
__global__ __launch_bounds__(256) void out_gemm_kernel(
    const _Float16* __restrict__ A, const float* __restrict__ W,
    const float* __restrict__ bias, float* __restrict__ out) {
  __shared__ __align__(32) _Float16 As[2][128 * 64];
  __shared__ __align__(32) _Float16 Bs[2][128 * 64];

  const int m0 = blockIdx.x * 128;
  const int n0 = blockIdx.y * 128;
  const int t = threadIdx.x;
  const int wave = t >> 5, lane = t & 31;
  const int mw = (wave & 3) * 32;
  const int nw = (wave >> 2) * 64;

  const int sr = t >> 1, sc = (t & 1) * 32;

  v8h a16[4];
  _Float16 b16[32];

  auto load_regs = [&](int k0) {
    const _Float16* ap = A + (size_t)(m0 + sr) * D_MODEL + k0 + sc;
    const float* bp = W + (size_t)(n0 + sr) * D_MODEL + k0 + sc;
#pragma unroll
    for (int i = 0; i < 4; ++i) a16[i] = *(const v8h*)(ap + i * 8);
#pragma unroll
    for (int i = 0; i < 32; ++i) b16[i] = (_Float16)bp[i];
  };
  auto store_stage = [&](int buf) {
    _Float16* ad = As[buf] + sr * 64 + sc;
    _Float16* bd = Bs[buf] + sr * 64 + sc;
#pragma unroll
    for (int i = 0; i < 4; ++i) *(v8h*)(ad + i * 8) = a16[i];
#pragma unroll
    for (int i = 0; i < 32; ++i) bd[i] = b16[i];
  };

  v8f acc[2][4] = {{{}, {}, {}, {}}, {{}, {}, {}, {}}};

  const int NS = D_MODEL / 64;
  load_regs(0);
  store_stage(0);
  for (int s = 0; s < NS; ++s) {
    __syncthreads();
    if (s + 1 < NS) load_regs((s + 1) * 64);

    const _Float16* Ab = As[s & 1] + mw * 64;
    v16h fa[2][2];
#pragma unroll
    for (int m = 0; m < 2; ++m) {
      fa[m][0] = load_frag_a(Ab + (16 * m) * 64, 64);
      fa[m][1] = load_frag_a(Ab + (16 * m) * 64 + 32, 64);
    }
#pragma unroll
    for (int j = 0; j < 4; ++j) {
      const _Float16* Bb = Bs[s & 1] + (nw + 16 * j) * 64;
      v16h fb0 = load_frag_b(Bb, 64);
      v16h fb1 = load_frag_b(Bb + 32, 64);
#pragma unroll
      for (int m = 0; m < 2; ++m) {
        acc[m][j] = wmma16x16x32(fa[m][0], fb0, acc[m][j]);
        acc[m][j] = wmma16x16x32(fa[m][1], fb1, acc[m][j]);
      }
    }
    if (s + 1 < NS) store_stage((s + 1) & 1);
  }

  const int hl = lane >> 4;
  const int nn = lane & 15;
#pragma unroll
  for (int m = 0; m < 2; ++m) {
#pragma unroll
    for (int j = 0; j < 4; ++j) {
#pragma unroll
      for (int r = 0; r < 8; ++r) {
        int grow = m0 + mw + 16 * m + r + 8 * hl;
        int gcol = n0 + nw + 16 * j + nn;
        __builtin_nontemporal_store(acc[m][j][r] + bias[gcol],
                                    out + (size_t)grow * D_MODEL + gcol);
      }
    }
  }
}

extern "C" void kernel_launch(void* const* d_in, const int* in_sizes, int n_in,
                              void* d_out, int out_size, void* d_ws, size_t ws_size,
                              hipStream_t stream) {
  const float* hidden = (const float*)d_in[0];   // [B,S,D]
  const float* mask   = (const float*)d_in[1];   // [B,1,1,S]
  const float* rel    = (const float*)d_in[2];   // [1,H,S,S]
  const float* Wq = (const float*)d_in[3];
  const float* bq = (const float*)d_in[4];
  const float* Wk = (const float*)d_in[5];
  const float* bk = (const float*)d_in[6];
  const float* Wv = (const float*)d_in[7];
  const float* bv = (const float*)d_in[8];
  const float* Wo = (const float*)d_in[9];
  const float* bo = (const float*)d_in[10];
  float* out = (float*)d_out;

  const size_t elems = (size_t)BATCH * S_LEN * D_MODEL;  // 4M
  _Float16* qws = (_Float16*)d_ws;
  _Float16* kws = qws + elems;
  _Float16* vws = kws + elems;
  _Float16* cws = vws + elems;   // 32 MB f16 scratch total

  dim3 gG(BATCH * S_LEN / 128, D_MODEL / 128, 1);  // (32, 8)
  qkv_gemm_kernel<<<gG, 256, 0, stream>>>(hidden, Wq, bq, qws, 0.125f); // 1/sqrt(64)
  qkv_gemm_kernel<<<gG, 256, 0, stream>>>(hidden, Wk, bk, kws, 1.0f);
  qkv_gemm_kernel<<<gG, 256, 0, stream>>>(hidden, Wv, bv, vws, 1.0f);

  dim3 gA(S_LEN / 128, NH, BATCH);                 // (16, 16, 2)
  attn_flash_kernel<<<gA, 256, 0, stream>>>(qws, kws, vws, rel, mask, cws);

  out_gemm_kernel<<<gG, 256, 0, stream>>>(cws, Wo, bo, out);
}
